// GCN_9689446219915
// MI455X (gfx1250) — compile-verified, matrix operand order checked by reading, not run on hardware
//
#include <hip/hip_runtime.h>
#include <hip/hip_bf16.h>

#define IN_F 512
#define HID  128
#define NCLS 40
#define NCLS_PAD 48

typedef __attribute__((ext_vector_type(16))) __bf16 v16bf;
typedef __attribute__((ext_vector_type(8)))  float  v8f;

union Frag {
    uint4 q[2];
    v16bf v;
};

__device__ __forceinline__ unsigned short f2bf(float f) {
    unsigned int u = __float_as_uint(f);
    u += 0x7fffu + ((u >> 16) & 1u);   // round-to-nearest-even
    return (unsigned short)(u >> 16);
}

// ---------------- weight pre-transpose (f32 [K][N] -> bf16 [N][K]) ----------------
__global__ void k_w1t(const float* __restrict__ W1, unsigned short* __restrict__ W1t) {
    int idx = blockIdx.x * blockDim.x + threadIdx.x;     // [HID * IN_F)
    if (idx >= HID * IN_F) return;
    int n = idx / IN_F, k = idx % IN_F;
    W1t[idx] = f2bf(W1[k * HID + n]);
}

__global__ void k_w2t(const float* __restrict__ W2, unsigned short* __restrict__ W2t) {
    int idx = blockIdx.x * blockDim.x + threadIdx.x;     // [NCLS_PAD * HID)
    if (idx >= NCLS_PAD * HID) return;
    int n = idx / HID, k = idx % HID;
    W2t[idx] = (n < NCLS) ? f2bf(W2[k * NCLS + n]) : (unsigned short)0;
}

// ---------------- degree / norm prep ----------------
__global__ void k_deg_init(float* __restrict__ deg, int n) {
    int i = blockIdx.x * blockDim.x + threadIdx.x;
    if (i < n) deg[i] = 1.0f;                            // self loop
}

__global__ void k_deg_acc(const int* __restrict__ dst, float* __restrict__ deg, int e) {
    int i = blockIdx.x * blockDim.x + threadIdx.x;
    if (i < e) atomicAdd(&deg[dst[i]], 1.0f);
}

__global__ void k_dinv(float* __restrict__ deg, int n) {
    int i = blockIdx.x * blockDim.x + threadIdx.x;
    if (i < n) deg[i] = rsqrtf(deg[i]);                  // deg >= 1 always
}

__global__ void k_norm(const int* __restrict__ src, const int* __restrict__ dst,
                       const float* __restrict__ dinv, float* __restrict__ norm, int e) {
    int i = blockIdx.x * blockDim.x + threadIdx.x;
    if (i < e) norm[i] = dinv[src[i]] * dinv[dst[i]];
}

// ---------------- GEMM1: h1 = x @ W1  (M=n, K=512, N=128), bf16 WMMA ----------------
// 256 threads = 8 waves; block -> 16 rows; wave w -> N-tile w (16 cols)
__global__ void __launch_bounds__(256) k_gemm1(const float* __restrict__ x,
                                               const unsigned short* __restrict__ W1t,
                                               float* __restrict__ h1) {
    __shared__ __align__(16) unsigned short As[16 * IN_F];   // 16 KB bf16 A tile
    const int t = threadIdx.x;
    const int row0 = blockIdx.x * 16;

    #pragma unroll
    for (int i = 0; i < 32; ++i) {                       // 8192 elems / 256 thr
        int idx = i * 256 + t;
        As[idx] = f2bf(x[(size_t)row0 * IN_F + idx]);
    }
    __syncthreads();

    const int wave = t >> 5;
    const int lane = t & 31;
    const int half = lane >> 4;          // 0: K+0, 1: K+8 sub-block
    const int l16  = lane & 15;
    const int koff = half * 8;
    const unsigned short* Bbase = W1t + (size_t)(wave * 16 + l16) * IN_F;

    v8f c = {};
    #pragma unroll
    for (int kk = 0; kk < IN_F; kk += 32) {
        Frag a, b;
        const unsigned short* ap = &As[l16 * IN_F + kk + koff];
        a.q[0] = *(const uint4*)(ap);
        a.q[1] = *(const uint4*)(ap + 16);
        const unsigned short* bp = Bbase + kk + koff;
        b.q[0] = *(const uint4*)(bp);
        b.q[1] = *(const uint4*)(bp + 16);
        c = __builtin_amdgcn_wmma_f32_16x16x32_bf16(false, a.v, false, b.v,
                                                    (short)0, c, false, false);
    }

    const int col = wave * 16 + l16;
    #pragma unroll
    for (int r = 0; r < 8; ++r) {
        int row = row0 + r + half * 8;
        h1[(size_t)row * HID + col] = c[r];
    }
}

// ---------------- self-loop init: agg1 = h1 * dinv^2 ----------------
__global__ void k_self1(const float* __restrict__ h1, const float* __restrict__ dinv,
                        float* __restrict__ agg1, int total) {
    int idx = blockIdx.x * blockDim.x + threadIdx.x;
    if (idx >= total) return;
    float d = dinv[idx >> 7];            // row = idx / 128
    agg1[idx] = h1[idx] * d * d;
}

// ---------------- edge scatter layer 1: one wave per edge, float4 per lane ----------------
__global__ void __launch_bounds__(256) k_scatter1(const int* __restrict__ src,
                                                  const int* __restrict__ dst,
                                                  const float* __restrict__ norm,
                                                  const float* __restrict__ h1,
                                                  float* __restrict__ agg1, int e) {
    int wid  = blockIdx.x * 8 + (threadIdx.x >> 5);
    int lane = threadIdx.x & 31;
    if (wid >= e) return;
    int s = src[wid], d = dst[wid];
    float w = norm[wid];
    float4 v = ((const float4*)(h1 + (size_t)s * HID))[lane];
    float* o = agg1 + (size_t)d * HID + lane * 4;
    atomicAdd(o + 0, v.x * w);
    atomicAdd(o + 1, v.y * w);
    atomicAdd(o + 2, v.z * w);
    atomicAdd(o + 3, v.w * w);
}

// ---------------- GEMM2: h2 = relu(agg1 + b1) @ W2  (K=128, N=40 padded to 48) ----------------
// 96 threads = 3 waves; block -> 16 rows; wave w -> N-tile w
__global__ void __launch_bounds__(96) k_gemm2(const float* __restrict__ agg1,
                                              const float* __restrict__ b1,
                                              const unsigned short* __restrict__ W2t,
                                              float* __restrict__ h2) {
    __shared__ __align__(16) unsigned short As[16 * HID];    // 4 KB
    const int t = threadIdx.x;
    const int row0 = blockIdx.x * 16;

    for (int idx = t; idx < 16 * HID; idx += 96) {
        float v = agg1[(size_t)row0 * HID + idx] + b1[idx & (HID - 1)];
        As[idx] = f2bf(fmaxf(v, 0.0f));                  // fused bias + ReLU
    }
    __syncthreads();

    const int wave = t >> 5;
    const int lane = t & 31;
    const int half = lane >> 4;
    const int l16  = lane & 15;
    const int koff = half * 8;
    const unsigned short* Bbase = W2t + (size_t)(wave * 16 + l16) * HID;

    v8f c = {};
    #pragma unroll
    for (int kk = 0; kk < HID; kk += 32) {
        Frag a, b;
        const unsigned short* ap = &As[l16 * HID + kk + koff];
        a.q[0] = *(const uint4*)(ap);
        a.q[1] = *(const uint4*)(ap + 16);
        const unsigned short* bp = Bbase + kk + koff;
        b.q[0] = *(const uint4*)(bp);
        b.q[1] = *(const uint4*)(bp + 16);
        c = __builtin_amdgcn_wmma_f32_16x16x32_bf16(false, a.v, false, b.v,
                                                    (short)0, c, false, false);
    }

    const int col = wave * 16 + l16;
    if (col < NCLS) {
        #pragma unroll
        for (int r = 0; r < 8; ++r) {
            int row = row0 + r + half * 8;
            h2[(size_t)row * NCLS + col] = c[r];
        }
    }
}

// ---------------- self-loop init + bias: out = h2 * dinv^2 + b2 ----------------
__global__ void k_self2(const float* __restrict__ h2, const float* __restrict__ dinv,
                        const float* __restrict__ b2, float* __restrict__ out, int total) {
    int idx = blockIdx.x * blockDim.x + threadIdx.x;
    if (idx >= total) return;
    int row = idx / NCLS, col = idx - row * NCLS;
    float d = dinv[row];
    out[idx] = h2[idx] * d * d + b2[col];
}

// ---------------- edge scatter layer 2 ----------------
__global__ void __launch_bounds__(256) k_scatter2(const int* __restrict__ src,
                                                  const int* __restrict__ dst,
                                                  const float* __restrict__ norm,
                                                  const float* __restrict__ h2,
                                                  float* __restrict__ out, int e) {
    int wid  = blockIdx.x * 8 + (threadIdx.x >> 5);
    int lane = threadIdx.x & 31;
    if (wid >= e) return;
    int s = src[wid], d = dst[wid];
    float w = norm[wid];
    const float* hs = h2 + (size_t)s * NCLS;
    float* o = out + (size_t)d * NCLS;
    atomicAdd(o + lane, hs[lane] * w);                   // cols 0..31
    if (lane < NCLS - 32)                                // cols 32..39
        atomicAdd(o + 32 + lane, hs[32 + lane] * w);
}

extern "C" void kernel_launch(void* const* d_in, const int* in_sizes, int n_in,
                              void* d_out, int out_size, void* d_ws, size_t ws_size,
                              hipStream_t stream) {
    const float* x  = (const float*)d_in[0];
    const int*   ei = (const int*)d_in[1];
    const float* W1 = (const float*)d_in[2];
    const float* b1 = (const float*)d_in[3];
    const float* W2 = (const float*)d_in[4];
    const float* b2 = (const float*)d_in[5];
    float* out = (float*)d_out;

    const int n = in_sizes[0] / IN_F;   // 100000 (divisible by 16)
    const int e = in_sizes[1] / 2;      // 1600000
    const int* src = ei;
    const int* dst = ei + e;

    char* ws = (char*)d_ws;
    size_t off = 0;
    auto alloc = [&](size_t bytes) -> void* {
        void* p = ws + off;
        off += (bytes + 255) & ~size_t(255);
        return p;
    };
    unsigned short* w1t  = (unsigned short*)alloc((size_t)HID * IN_F * 2);
    unsigned short* w2t  = (unsigned short*)alloc((size_t)NCLS_PAD * HID * 2);
    float*          dinv = (float*)alloc((size_t)n * 4);          // deg -> dinv in place
    float*          norm = (float*)alloc((size_t)e * 4);
    float*          h1   = (float*)alloc((size_t)n * HID * 4);
    float*          agg1 = (float*)alloc((size_t)n * HID * 4);
    float*          h2   = (float*)alloc((size_t)n * NCLS * 4);

    // weight conversion + degree/norm prep
    k_w1t<<<(HID * IN_F + 255) / 256, 256, 0, stream>>>(W1, w1t);
    k_w2t<<<(NCLS_PAD * HID + 255) / 256, 256, 0, stream>>>(W2, w2t);
    k_deg_init<<<(n + 255) / 256, 256, 0, stream>>>(dinv, n);
    k_deg_acc<<<(e + 255) / 256, 256, 0, stream>>>(dst, dinv, e);
    k_dinv<<<(n + 255) / 256, 256, 0, stream>>>(dinv, n);
    k_norm<<<(e + 255) / 256, 256, 0, stream>>>(src, dst, dinv, norm, e);

    // layer 1
    k_gemm1<<<n / 16, 256, 0, stream>>>(x, w1t, h1);
    k_self1<<<(n * HID + 255) / 256, 256, 0, stream>>>(h1, dinv, agg1, n * HID);
    k_scatter1<<<(e + 7) / 8, 256, 0, stream>>>(src, dst, norm, h1, agg1, e);

    // layer 2
    k_gemm2<<<n / 16, 96, 0, stream>>>(agg1, b1, w2t, h2);
    k_self2<<<(n * NCLS + 255) / 256, 256, 0, stream>>>(h2, dinv, b2, out, n * NCLS);
    k_scatter2<<<(e + 7) / 8, 256, 0, stream>>>(src, dst, norm, h2, out, e);
}
